// CrossDeepNetwork_59004260713083
// MI455X (gfx1250) — matrix operand order repaired and checked by reading, not compiled, not run on hardware
//
#include <hip/hip_runtime.h>
#include <math.h>

#define BATCH   4096
#define NSP     26
#define VOCABSZ 100000
#define EMBD    106
#define NDENSE  13
#define D_RAW   2769          // 26*106 + 13
#define D_PAD   2784          // next multiple of 32
#define NCROSS  4
#define H1N     1024
#define H2N     1024
#define H3N     512
#define BN_EPS  1e-5f
#define CHUNKS  16

typedef __attribute__((ext_vector_type(16))) __bf16 v16bf;
typedef __attribute__((ext_vector_type(8)))  float  v8f;

struct Pair128 { uint4 a, b; };

__device__ __forceinline__ v16bf make_frag(uint4 a, uint4 b) {
  Pair128 p{a, b};
  return __builtin_bit_cast(v16bf, p);
}

// Split f32 into bf16 hi (truncation) + bf16 lo (residual, truncated).
__device__ __forceinline__ void split_bf16(float v, unsigned short& h, unsigned short& l) {
  unsigned hu = __builtin_bit_cast(unsigned, v) & 0xFFFF0000u;
  float hf = __builtin_bit_cast(float, hu);
  float lf = v - hf;
  h = (unsigned short)(hu >> 16);
  l = (unsigned short)(__builtin_bit_cast(unsigned, lf) >> 16);
}

// CDNA5 async global->LDS copy (VGLOBAL op 98, tracked by ASYNCcnt).
// lds_off: byte offset into wave's LDS (low 32 bits of generic shared ptr).
__device__ __forceinline__ void async_load_b128(unsigned lds_off, const void* gptr) {
  asm volatile("global_load_async_to_lds_b128 %0, %1, off"
               :: "v"(lds_off), "v"(gptr)
               : "memory");
}

__device__ __forceinline__ unsigned lds_addr(const void* p) {
  return (unsigned)(uintptr_t)p;   // low 32 bits = LDS offset (ISA 10.2)
}

// ---------------------------------------------------------------------------
// 1) Embedding gather + dense concat -> X0 raw [BATCH][D_PAD], zero padded.
// ---------------------------------------------------------------------------
__global__ __launch_bounds__(256)
void k_gather_concat(const int* __restrict__ sparse,
                     const float* __restrict__ dense,
                     const float* __restrict__ emb,
                     float* __restrict__ X0) {
  const int r = blockIdx.x;
  __shared__ int sidx[NSP];
  if (threadIdx.x < NSP) sidx[threadIdx.x] = sparse[r * NSP + threadIdx.x];
  __syncthreads();
  for (int d = threadIdx.x; d < D_PAD; d += blockDim.x) {
    float v;
    if (d < NSP * EMBD) {
      int f = d / EMBD;
      int e = d - f * EMBD;
      v = emb[((size_t)f * VOCABSZ + sidx[f]) * EMBD + e];
    } else if (d < D_RAW) {
      v = dense[r * NDENSE + (d - NSP * EMBD)];
    } else {
      v = 0.0f;
    }
    X0[(size_t)r * D_PAD + d] = v;
  }
}

// ---------------------------------------------------------------------------
// 2) Column statistics (two-phase); BN apply; BN apply + bf16 split.
// ---------------------------------------------------------------------------
__global__ void k_col_partial(const float* __restrict__ X,
                              float* __restrict__ psum, float* __restrict__ psq,
                              int R, int C) {
  const int c = blockIdx.x * blockDim.x + threadIdx.x;
  const int chunk = blockIdx.y;
  if (c >= C) return;
  const int rows = R / CHUNKS;
  const int r0 = chunk * rows;
  float s = 0.0f, q = 0.0f;
  for (int r = r0; r < r0 + rows; ++r) {
    float v = X[(size_t)r * C + c];
    s += v;
    q += v * v;
  }
  psum[(size_t)chunk * C + c] = s;
  psq [(size_t)chunk * C + c] = q;
}

__global__ void k_col_finalize(const float* __restrict__ psum,
                               const float* __restrict__ psq,
                               float* __restrict__ mean, float* __restrict__ rsig,
                               int R, int C) {
  const int c = blockIdx.x * blockDim.x + threadIdx.x;
  if (c >= C) return;
  float s = 0.0f, q = 0.0f;
  for (int k = 0; k < CHUNKS; ++k) {
    s += psum[(size_t)k * C + c];
    q += psq [(size_t)k * C + c];
  }
  const float inv = 1.0f / (float)R;
  const float mu = s * inv;
  const float var = q * inv - mu * mu;   // biased variance (matches jnp.var)
  mean[c] = mu;
  rsig[c] = rsqrtf(var + BN_EPS);
}

__global__ void k_bn_apply(float* __restrict__ X,
                           const float* __restrict__ mean,
                           const float* __restrict__ rsig,
                           int total, int C, int relu) {
  const int i = blockIdx.x * blockDim.x + threadIdx.x;
  if (i >= total) return;
  const int c = i % C;
  float v = (X[i] - mean[c]) * rsig[c];
  if (relu) v = fmaxf(v, 0.0f);
  X[i] = v;
}

__global__ void k_bn_split(float* __restrict__ X,
                           const float* __restrict__ mean,
                           const float* __restrict__ rsig,
                           unsigned short* __restrict__ Xh,
                           unsigned short* __restrict__ Xl,
                           int total, int C, int relu) {
  const int i = blockIdx.x * blockDim.x + threadIdx.x;
  if (i >= total) return;
  const int c = i % C;
  float v = (X[i] - mean[c]) * rsig[c];
  if (relu) v = fmaxf(v, 0.0f);
  X[i] = v;
  unsigned short h, l;
  split_bf16(v, h, l);
  Xh[i] = h;
  Xl[i] = l;
}

// ---------------------------------------------------------------------------
// 3) Cross network: x_{i+1} = x0 * (x_i . w_i) + b_i + x_i  (4 layers)
// ---------------------------------------------------------------------------
__global__ __launch_bounds__(256)
void k_cross(const float* __restrict__ X0,       // [BATCH][D_PAD] (normalized)
             const float* __restrict__ cw,       // [NCROSS][D_RAW]
             const float* __restrict__ cb,       // [NCROSS][D_RAW]
             float* __restrict__ Xc) {           // [BATCH][D_RAW]
  __shared__ float x0s[D_RAW];
  __shared__ float xs[D_RAW];
  __shared__ float red[256];
  const int r = blockIdx.x;
  const int tid = threadIdx.x;
  const float* xr = X0 + (size_t)r * D_PAD;
  for (int d = tid; d < D_RAW; d += 256) {
    float v = xr[d];
    x0s[d] = v;
    xs[d] = v;
  }
  __syncthreads();
  for (int i = 0; i < NCROSS; ++i) {
    float p = 0.0f;
    for (int d = tid; d < D_RAW; d += 256) p += xs[d] * cw[i * D_RAW + d];
    red[tid] = p;
    __syncthreads();
    for (int s = 128; s > 0; s >>= 1) {
      if (tid < s) red[tid] += red[tid + s];
      __syncthreads();
    }
    const float dot = red[0];
    for (int d = tid; d < D_RAW; d += 256)
      xs[d] = x0s[d] * dot + cb[i * D_RAW + d] + xs[d];
    __syncthreads();
  }
  for (int d = tid; d < D_RAW; d += 256) Xc[(size_t)r * D_RAW + d] = xs[d];
}

// ---------------------------------------------------------------------------
// 4) Weight transpose + pad + split: W[Kin][N] f32 -> Wt_hi/lo[N][Kp] bf16.
// ---------------------------------------------------------------------------
__global__ __launch_bounds__(256)
void k_wsplit_t(const float* __restrict__ W,
                unsigned short* __restrict__ Th,
                unsigned short* __restrict__ Tl,
                int Kin, int N, int Kp) {
  __shared__ float tile[32][33];
  const int kb = blockIdx.x * 32;
  const int nb = blockIdx.y * 32;
  const int tx = threadIdx.x & 31;
  const int ty = threadIdx.x >> 5;    // 0..7
  #pragma unroll
  for (int i = 0; i < 4; ++i) {
    const int k = kb + ty + i * 8;
    const int n = nb + tx;
    float v = 0.0f;
    if (k < Kin && n < N) v = W[(size_t)k * N + n];
    tile[ty + i * 8][tx] = v;
  }
  __syncthreads();
  #pragma unroll
  for (int i = 0; i < 4; ++i) {
    const int n = nb + ty + i * 8;
    const int k = kb + tx;
    if (n < N && k < Kp) {
      unsigned short h, l;
      split_bf16(tile[tx][ty + i * 8], h, l);
      Th[(size_t)n * Kp + k] = h;
      Tl[(size_t)n * Kp + k] = l;
    }
  }
}

// ---------------------------------------------------------------------------
// 5) Split-bf16 GEMM (3 products): C = (Ah+Al)(Bh+Bl)^T ~ hh + hl + lh,
//    fp32 accumulation via v_wmma_f32_16x16x32_bf16.
//    Double-buffered LDS fed by GLOBAL_LOAD_ASYNC_TO_LDS_B128 (ASYNCcnt).
//    Block 256 thr = 8 waves; tile 128x64; wave tile 32x32; K-block 32.
//    Requires M%128==0, N%64==0, K%32==0.
// ---------------------------------------------------------------------------
#define GBM 128
#define GBN 64
#define GKB 32
#define LDS_S 40   // ushort row stride: 80B (16B aligned, conflict-free b128)

__global__ __launch_bounds__(256)
void k_gemm_bf16x3(const unsigned short* __restrict__ Ah,   // [M][K]
                   const unsigned short* __restrict__ Al,
                   const unsigned short* __restrict__ Bth,  // [N][K]
                   const unsigned short* __restrict__ Btl,
                   float* __restrict__ C,                   // [M][N]
                   int M, int N, int K) {
  __shared__ unsigned short AsH[2][GBM * LDS_S];
  __shared__ unsigned short AsL[2][GBM * LDS_S];
  __shared__ unsigned short BsH[2][GBN * LDS_S];
  __shared__ unsigned short BsL[2][GBN * LDS_S];

  const int tid   = threadIdx.x;
  const int lane  = tid & 31;
  const int wave  = tid >> 5;
  const int l16   = lane & 15;
  const int lhalf = lane >> 4;
  const int waveM = wave >> 1;        // 0..3
  const int waveN = wave & 1;         // 0..1
  const size_t blockM = (size_t)blockIdx.x * GBM;
  const size_t blockN = (size_t)blockIdx.y * GBN;

  // staging geometry: A tile 128x32 (512 xb128, 2/thread), B tile 64x32 (1/thread)
  const int aRow0 = tid >> 2;               // 0..63
  const int aRow1 = (tid + 256) >> 2;       // 64..127
  const int aC0 = (tid & 3) * 8;
  const int bRow = tid >> 2;                // 0..63
  const int bC = (tid & 3) * 8;
  const size_t aOff0 = (blockM + aRow0) * (size_t)K;
  const size_t aOff1 = (blockM + aRow1) * (size_t)K;
  const size_t bOff  = (blockN + bRow) * (size_t)K;

  // per-buffer LDS byte addresses for this thread's staging slots
  unsigned ldsAh[2], ldsAh1[2], ldsAl[2], ldsAl1[2], ldsBh[2], ldsBl[2];
  #pragma unroll
  for (int b = 0; b < 2; ++b) {
    ldsAh [b] = lds_addr(&AsH[b][aRow0 * LDS_S + aC0]);
    ldsAh1[b] = lds_addr(&AsH[b][aRow1 * LDS_S + aC0]);
    ldsAl [b] = lds_addr(&AsL[b][aRow0 * LDS_S + aC0]);
    ldsAl1[b] = lds_addr(&AsL[b][aRow1 * LDS_S + aC0]);
    ldsBh [b] = lds_addr(&BsH[b][bRow * LDS_S + bC]);
    ldsBl [b] = lds_addr(&BsL[b][bRow * LDS_S + bC]);
  }

  auto issue_group = [&](int buf, int k0) {
    async_load_b128(ldsAh [buf], Ah  + aOff0 + k0 + aC0);
    async_load_b128(ldsAh1[buf], Ah  + aOff1 + k0 + aC0);
    async_load_b128(ldsAl [buf], Al  + aOff0 + k0 + aC0);
    async_load_b128(ldsAl1[buf], Al  + aOff1 + k0 + aC0);
    async_load_b128(ldsBh [buf], Bth + bOff  + k0 + bC);
    async_load_b128(ldsBl [buf], Btl + bOff  + k0 + bC);
  };

  // fragment geometry (ISA 7.12.2):
  // A 16x32 16-bit: lane<16 -> M=l16, K-chunks {0..7}+{16..23}; lane>=16 -> +8
  // B 32x16 16-bit: lane<16 -> N=l16, K-chunks {0..7}+{16..23}; lane>=16 -> +8
  const int amOff[2] = { (waveM * 32 + l16) * LDS_S,
                         (waveM * 32 + 16 + l16) * LDS_S };
  const int bnOff[2] = { (waveN * 32 + l16) * LDS_S,
                         (waveN * 32 + 16 + l16) * LDS_S };
  const int aK1 = 8 * lhalf;
  const int aK2 = 16 + 8 * lhalf;
  const int bK1 = 8 * lhalf;
  const int bK2 = 16 + 8 * lhalf;

  v8f acc[2][2] = {};

  const int numKB = K / GKB;
  issue_group(0, 0);
  if (numKB > 1) issue_group(1, GKB);

  for (int kb = 0; kb < numKB; ++kb) {
    // wait until group kb has fully landed in LDS (one newer group may remain)
    if (kb + 1 < numKB) asm volatile("s_wait_asynccnt 0x6" ::: "memory");
    else                asm volatile("s_wait_asynccnt 0x0" ::: "memory");
    __syncthreads();

    const int buf = kb & 1;
    const unsigned short* ah = AsH[buf];
    const unsigned short* al = AsL[buf];
    const unsigned short* bh = BsH[buf];
    const unsigned short* bl = BsL[buf];

    v16bf fAh[2], fAl[2], fBh[2], fBl[2];
    #pragma unroll
    for (int mt = 0; mt < 2; ++mt) {
      fAh[mt] = make_frag(*reinterpret_cast<const uint4*>(&ah[amOff[mt] + aK1]),
                          *reinterpret_cast<const uint4*>(&ah[amOff[mt] + aK2]));
      fAl[mt] = make_frag(*reinterpret_cast<const uint4*>(&al[amOff[mt] + aK1]),
                          *reinterpret_cast<const uint4*>(&al[amOff[mt] + aK2]));
    }
    #pragma unroll
    for (int nt = 0; nt < 2; ++nt) {
      fBh[nt] = make_frag(*reinterpret_cast<const uint4*>(&bh[bnOff[nt] + bK1]),
                          *reinterpret_cast<const uint4*>(&bh[bnOff[nt] + bK2]));
      fBl[nt] = make_frag(*reinterpret_cast<const uint4*>(&bl[bnOff[nt] + bK1]),
                          *reinterpret_cast<const uint4*>(&bl[bnOff[nt] + bK2]));
    }
    #pragma unroll
    for (int mt = 0; mt < 2; ++mt) {
      #pragma unroll
      for (int nt = 0; nt < 2; ++nt) {
        acc[mt][nt] = __builtin_amdgcn_wmma_f32_16x16x32_bf16(
            false, fAh[mt], false, fBh[nt], (short)0, acc[mt][nt], false, false);
        acc[mt][nt] = __builtin_amdgcn_wmma_f32_16x16x32_bf16(
            false, fAh[mt], false, fBl[nt], (short)0, acc[mt][nt], false, false);
        acc[mt][nt] = __builtin_amdgcn_wmma_f32_16x16x32_bf16(
            false, fAl[mt], false, fBh[nt], (short)0, acc[mt][nt], false, false);
      }
    }
    __syncthreads();   // all waves done reading buf before it is overwritten
    if (kb + 2 < numKB) issue_group(buf, (kb + 2) * GKB);
  }

  // writeback: C/D 16x16 f32: VGPR r, lanes 0-15 -> M=r, lanes 16-31 -> M=r+8
  #pragma unroll
  for (int mt = 0; mt < 2; ++mt) {
    #pragma unroll
    for (int nt = 0; nt < 2; ++nt) {
      const size_t col = blockN + waveN * 32 + nt * 16 + l16;
      #pragma unroll
      for (int r = 0; r < 8; ++r) {
        const size_t row = blockM + waveM * 32 + mt * 16 + r + 8 * lhalf;
        C[row * N + col] = acc[mt][nt][r];
      }
    }
  }
}

// ---------------------------------------------------------------------------
// 6) Final: sigmoid( [x_cross || h3] . out_w ), one block per row.
// ---------------------------------------------------------------------------
__global__ __launch_bounds__(256)
void k_final(const float* __restrict__ Xc,     // [BATCH][D_RAW]
             const float* __restrict__ H3,     // [BATCH][H3N]
             const float* __restrict__ ow,     // [D_RAW + H3N]
             float* __restrict__ out) {
  __shared__ float red[256];
  const int r = blockIdx.x;
  const int tid = threadIdx.x;
  float p = 0.0f;
  for (int d = tid; d < D_RAW; d += 256) p += Xc[(size_t)r * D_RAW + d] * ow[d];
  for (int j = tid; j < H3N; j += 256) p += H3[(size_t)r * H3N + j] * ow[D_RAW + j];
  red[tid] = p;
  __syncthreads();
  for (int s = 128; s > 0; s >>= 1) {
    if (tid < s) red[tid] += red[tid + s];
    __syncthreads();
  }
  if (tid == 0) out[r] = 1.0f / (1.0f + expf(-red[0]));
}

// ---------------------------------------------------------------------------
// Launcher
// ---------------------------------------------------------------------------
extern "C" void kernel_launch(void* const* d_in, const int* in_sizes, int n_in,
                              void* d_out, int out_size, void* d_ws, size_t ws_size,
                              hipStream_t stream) {
  const int*   sparse = (const int*)  d_in[0];
  const float* dense  = (const float*)d_in[1];
  const float* emb    = (const float*)d_in[2];
  const float* cw     = (const float*)d_in[3];
  const float* cb     = (const float*)d_in[4];
  const float* w1     = (const float*)d_in[5];
  // d_in[6] deep_b1: zeros, cancelled by affine-free BN -- unused
  const float* w2     = (const float*)d_in[7];
  // d_in[8] deep_b2 -- unused
  const float* w3     = (const float*)d_in[9];
  // d_in[10] deep_b3 -- unused
  const float* ow     = (const float*)d_in[11];
  float* out = (float*)d_out;

  // Workspace carve-up (256B-aligned chunks)
  char* base = (char*)d_ws;
  auto carve = [&](size_t bytes) -> char* {
    char* p = base;
    base += (bytes + 255) & ~(size_t)255;
    return p;
  };
  float* X0   = (float*)carve((size_t)BATCH * D_PAD * 4);
  float* Xc   = (float*)carve((size_t)BATCH * D_RAW * 4);
  float* H1   = (float*)carve((size_t)BATCH * H1N * 4);
  float* H2   = (float*)carve((size_t)BATCH * H2N * 4);
  float* H3   = (float*)carve((size_t)BATCH * H3N * 4);
  unsigned short* Ah  = (unsigned short*)carve((size_t)BATCH * D_PAD * 2); // reused per layer
  unsigned short* Al  = (unsigned short*)carve((size_t)BATCH * D_PAD * 2);
  unsigned short* W1h = (unsigned short*)carve((size_t)H1N * D_PAD * 2);
  unsigned short* W1l = (unsigned short*)carve((size_t)H1N * D_PAD * 2);
  unsigned short* W2h = (unsigned short*)carve((size_t)H2N * H1N * 2);
  unsigned short* W2l = (unsigned short*)carve((size_t)H2N * H1N * 2);
  unsigned short* W3h = (unsigned short*)carve((size_t)H3N * H2N * 2);
  unsigned short* W3l = (unsigned short*)carve((size_t)H3N * H2N * 2);
  float* psum = (float*)carve((size_t)CHUNKS * D_PAD * 4);
  float* psq  = (float*)carve((size_t)CHUNKS * D_PAD * 4);
  float* mean = (float*)carve((size_t)D_PAD * 4);
  float* rsig = (float*)carve((size_t)D_PAD * 4);

  auto run_stats = [&](float* X, int R, int C) {
    dim3 gp((C + 255) / 256, CHUNKS);
    k_col_partial<<<gp, 256, 0, stream>>>(X, psum, psq, R, C);
    k_col_finalize<<<(C + 255) / 256, 256, 0, stream>>>(psum, psq, mean, rsig, R, C);
  };

  // 1) gather + concat (zero-padded to D_PAD)
  k_gather_concat<<<BATCH, 256, 0, stream>>>(sparse, dense, emb, X0);

  // 2) BN(x0) in place + bf16 split (pad columns stay exactly zero)
  run_stats(X0, BATCH, D_PAD);
  {
    int total = BATCH * D_PAD;
    k_bn_split<<<(total + 255) / 256, 256, 0, stream>>>(
        X0, mean, rsig, Ah, Al, total, D_PAD, /*relu=*/0);
  }

  // 3) cross network
  k_cross<<<BATCH, 256, 0, stream>>>(X0, cw, cb, Xc);

  // 4) weight transpose + pad + split (once per launch)
  k_wsplit_t<<<dim3(D_PAD / 32, H1N / 32), 256, 0, stream>>>(w1, W1h, W1l, D_RAW, H1N, D_PAD);
  k_wsplit_t<<<dim3(H1N / 32, H2N / 32), 256, 0, stream>>>(w2, W2h, W2l, H1N, H2N, H1N);
  k_wsplit_t<<<dim3(H2N / 32, H3N / 32), 256, 0, stream>>>(w3, W3h, W3l, H2N, H3N, H2N);

  // 5) deep layer 1
  k_gemm_bf16x3<<<dim3(BATCH / GBM, H1N / GBN), 256, 0, stream>>>(
      Ah, Al, W1h, W1l, H1, BATCH, H1N, D_PAD);
  run_stats(H1, BATCH, H1N);
  {
    int total = BATCH * H1N;
    k_bn_split<<<(total + 255) / 256, 256, 0, stream>>>(
        H1, mean, rsig, Ah, Al, total, H1N, /*relu=*/1);
  }

  // 6) deep layer 2
  k_gemm_bf16x3<<<dim3(BATCH / GBM, H2N / GBN), 256, 0, stream>>>(
      Ah, Al, W2h, W2l, H2, BATCH, H2N, H1N);
  run_stats(H2, BATCH, H2N);
  {
    int total = BATCH * H2N;
    k_bn_split<<<(total + 255) / 256, 256, 0, stream>>>(
        H2, mean, rsig, Ah, Al, total, H2N, /*relu=*/1);
  }

  // 7) deep layer 3
  k_gemm_bf16x3<<<dim3(BATCH / GBM, H3N / GBN), 256, 0, stream>>>(
      Ah, Al, W3h, W3l, H3, BATCH, H3N, H2N);
  run_stats(H3, BATCH, H3N);
  {
    int total = BATCH * H3N;
    k_bn_apply<<<(total + 255) / 256, 256, 0, stream>>>(
        H3, mean, rsig, total, H3N, /*relu=*/1);
  }

  // 8) final concat-dot + sigmoid
  k_final<<<BATCH, 256, 0, stream>>>(Xc, H3, ow, out);
}
